// GatNet_90288802496748
// MI455X (gfx1250) — compile-verified, matrix operand order checked by reading, not run on hardware
//
#include <hip/hip_runtime.h>
#include <hip/hip_bf16.h>

// ---------------------------------------------------------------------------
// Types for WMMA
// ---------------------------------------------------------------------------
typedef __attribute__((ext_vector_type(16))) _Float16 v16h;
typedef __attribute__((ext_vector_type(8)))  _Float16 v8h;
typedef __attribute__((ext_vector_type(8)))  float    v8f;

#define NEG_SLOPE 0.2f
#define NGRAPHS   512

// ---------------------------------------------------------------------------
// Helpers
// ---------------------------------------------------------------------------
__device__ __forceinline__ void atomicMaxF(float* addr, float val) {
    // Works for any sign given init = -inf (0xFF800000)
    if (val >= 0.0f) {
        atomicMax((int*)addr, __float_as_int(val));
    } else {
        atomicMin((unsigned int*)addr, __float_as_uint(val));
    }
}

__device__ __forceinline__ void edge_nodes(const int* __restrict__ ei, int e, int NE,
                                           int& s, int& d) {
    if (e < NE) { s = ei[e]; d = ei[NE + e]; }
    else        { s = e - NE; d = e - NE; }     // self-loops appended
}

// ---------------------------------------------------------------------------
// Generic fill
// ---------------------------------------------------------------------------
__global__ void fill_f32_k(float* __restrict__ p, float v, int n) {
    int i = blockIdx.x * blockDim.x + threadIdx.x;
    if (i < n) p[i] = v;
}

// ---------------------------------------------------------------------------
// Layer 1 linear: h = x @ W1   (K = 2, M = 64) — scalar FMA is optimal here
// ---------------------------------------------------------------------------
__global__ void lin1_k(const float* __restrict__ x, const float* __restrict__ W1,
                       float* __restrict__ hlin, int n) {
    int i = blockIdx.x * blockDim.x + threadIdx.x;
    if (i >= n * 64) return;
    int nd = i >> 6, c = i & 63;
    hlin[i] = x[nd * 2] * W1[c] + x[nd * 2 + 1] * W1[64 + c];
}

// ---------------------------------------------------------------------------
// Pack W [K,128] f32 -> per-lane WMMA B-fragments (f16), exact ISA layout:
//   lane l, half i:  n = jt*16 + (l&15)
//                    k = kt*32 + (i>=8 ? 16:0) + (l>=16 ? 8:0) + (i&7)
// ---------------------------------------------------------------------------
template<int KT>
__global__ void pack_wfrag_k(const float* __restrict__ W, _Float16* __restrict__ Wfrag) {
    const int total = 8 * KT * 32 * 16;
    int tid = blockIdx.x * blockDim.x + threadIdx.x;
    if (tid >= total) return;
    int i    = tid & 15;
    int lane = (tid >> 4) & 31;
    int t    = tid >> 9;
    int kt   = t % KT;
    int jt   = t / KT;
    int n = jt * 16 + (lane & 15);
    int k = kt * 32 + ((i >= 8) ? 16 : 0) + ((lane >= 16) ? 8 : 0) + (i & 7);
    Wfrag[tid] = (_Float16)W[k * 128 + n];
}

// ---------------------------------------------------------------------------
// WMMA GEMM: out[N,128] f32 = Ah[N,K] f16 @ (packed W) f16
// 1 wave = 16 output rows; block = 4 waves = 64 rows.
// ---------------------------------------------------------------------------
template<int KT>
__global__ void gat_gemm_wmma_k(const _Float16* __restrict__ Ah,
                                const _Float16* __restrict__ Wfrag,
                                float* __restrict__ out, int nNodes) {
    const int K = KT * 32;
    int wave = threadIdx.x >> 5;
    int lane = threadIdx.x & 31;
    int row0 = (blockIdx.x * 4 + wave) * 16;
    if (row0 >= nNodes) return;                       // wave-uniform: EXEC stays all-ones

    int mrowlane = lane & 15;
    int grp      = (lane >> 4) * 8;                   // lane-group K offset
    const _Float16* abase = Ah + (size_t)(row0 + mrowlane) * K + grp;

    v16h afrag[KT];
#pragma unroll
    for (int kt = 0; kt < KT; ++kt) {
        v8h lo = *(const v8h*)(abase + kt * 32);       // K = kt*32+grp+{0..7}
        v8h hi = *(const v8h*)(abase + kt * 32 + 16);  // K = kt*32+grp+{16..23}
        afrag[kt] = __builtin_shufflevector(lo, hi, 0,1,2,3,4,5,6,7,8,9,10,11,12,13,14,15);
    }

    int col  = lane & 15;
    int mrow = row0 + grp;
#pragma unroll
    for (int jt = 0; jt < 8; ++jt) {
        v8f acc = {};
#pragma unroll
        for (int kt = 0; kt < KT; ++kt) {
            v16h b = *(const v16h*)(Wfrag + ((size_t)(jt * KT + kt) * 32 + lane) * 16);
            acc = __builtin_amdgcn_wmma_f32_16x16x32_f16(
                false, afrag[kt], false, b, (short)0, acc, false, false);
        }
#pragma unroll
        for (int v = 0; v < 8; ++v) {
            out[(size_t)(mrow + v) * 128 + jt * 16 + col] = acc[v];
        }
    }
}

// ---------------------------------------------------------------------------
// Per-node attention coefficients: es[n,h] = sum_c hlin[n,h*C+c]*a_s[h,c]
// ---------------------------------------------------------------------------
__global__ void attn_coef_k(const float* __restrict__ hlin,
                            const float* __restrict__ as_, const float* __restrict__ ad_,
                            float* __restrict__ es, float* __restrict__ ed,
                            int n, int C) {
    int i = blockIdx.x * blockDim.x + threadIdx.x;
    if (i >= n * 8) return;
    int h = i & 7, nd = i >> 3;
    const float* hp = hlin + (size_t)nd * (8 * C) + h * C;
    float s = 0.f, d = 0.f;
    for (int c = 0; c < C; ++c) {
        float v = hp[c];
        s += v * as_[h * C + c];
        d += v * ad_[h * C + c];
    }
    es[i] = s; ed[i] = d;
}

// ---------------------------------------------------------------------------
// Edge passes (segment softmax over dst)
// ---------------------------------------------------------------------------
__global__ void edge_max_k(const int* __restrict__ ei, const float* __restrict__ es,
                           const float* __restrict__ ed, float* __restrict__ mbuf,
                           int NE, int total) {
    int i = blockIdx.x * blockDim.x + threadIdx.x;
    if (i >= total) return;
    int e = i >> 3, h = i & 7, s, d;
    edge_nodes(ei, e, NE, s, d);
    float ev = es[s * 8 + h] + ed[d * 8 + h];
    ev = (ev > 0.f) ? ev : NEG_SLOPE * ev;
    atomicMaxF(&mbuf[d * 8 + h], ev);
}

__global__ void edge_expsum_k(const int* __restrict__ ei, const float* __restrict__ es,
                              const float* __restrict__ ed, const float* __restrict__ mbuf,
                              float* __restrict__ ssum, float* __restrict__ walpha,
                              int NE, int total) {
    int i = blockIdx.x * blockDim.x + threadIdx.x;
    if (i >= total) return;
    int e = i >> 3, h = i & 7, s, d;
    edge_nodes(ei, e, NE, s, d);
    float ev = es[s * 8 + h] + ed[d * 8 + h];
    ev = (ev > 0.f) ? ev : NEG_SLOPE * ev;
    float w = __expf(ev - mbuf[d * 8 + h]);
    walpha[i] = w;
    atomicAdd(&ssum[d * 8 + h], w);
}

__global__ void edge_alpha_k(const int* __restrict__ ei, const float* __restrict__ ssum,
                             float* __restrict__ walpha, int NE, int total) {
    int i = blockIdx.x * blockDim.x + threadIdx.x;
    if (i >= total) return;
    int e = i >> 3, h = i & 7, s, d;
    edge_nodes(ei, e, NE, s, d);
    walpha[i] = walpha[i] / (ssum[d * 8 + h] + 1e-16f);
}

// channel-parallel scatter: haggr[dst, ch] += hlin[src, ch] * alpha[e, ch>>cshift]
__global__ void edge_aggr_k(const int* __restrict__ ei, const float* __restrict__ hlin,
                            const float* __restrict__ walpha, float* __restrict__ haggr,
                            int NE, int hcshift, int cshift, int total) {
    int i = blockIdx.x * blockDim.x + threadIdx.x;
    if (i >= total) return;
    int HC = 1 << hcshift;
    int e  = i >> hcshift;
    int ch = i & (HC - 1);
    int s, d;
    edge_nodes(ei, e, NE, s, d);
    int h = ch >> cshift;
    float a = walpha[e * 8 + h];
    atomicAdd(&haggr[(size_t)d * HC + ch], hlin[(size_t)s * HC + ch] * a);
}

// ---------------------------------------------------------------------------
// Finalize: + bias, optional ELU, write f32 and/or f16 (next-layer WMMA input)
// ---------------------------------------------------------------------------
__global__ void gat_finalize_k(const float* __restrict__ haggr, const float* __restrict__ bias,
                               float* __restrict__ outF32, _Float16* __restrict__ outF16,
                               int HC, int total, int doElu) {
    int i = blockIdx.x * blockDim.x + threadIdx.x;
    if (i >= total) return;
    int c = i & (HC - 1);
    float v = haggr[i] + bias[c];
    if (doElu) v = (v > 0.f) ? v : expm1f(v);
    if (outF32) outF32[i] = v;
    if (outF16) outF16[i] = (_Float16)v;
}

// ---------------------------------------------------------------------------
// Global mean pool + MLP head + log_softmax
// ---------------------------------------------------------------------------
__global__ void pool_sum_k(const float* __restrict__ h3, const int* __restrict__ batch,
                           float* __restrict__ gsum, int total) {
    int i = blockIdx.x * blockDim.x + threadIdx.x;
    if (i >= total) return;
    int nd = i >> 7, c = i & 127;
    atomicAdd(&gsum[batch[nd] * 128 + c], h3[i]);
}

__global__ void pool_cnt_k(const int* __restrict__ batch, float* __restrict__ gcnt, int n) {
    int i = blockIdx.x * blockDim.x + threadIdx.x;
    if (i < n) atomicAdd(&gcnt[batch[i]], 1.0f);
}

__global__ void head_k(const float* __restrict__ gsum, const float* __restrict__ gcnt,
                       const float* __restrict__ fc1W, const float* __restrict__ fc1b,
                       const float* __restrict__ fc2W, const float* __restrict__ fc2b,
                       float* __restrict__ out) {
    __shared__ float pooled[128];
    __shared__ float z[32];
    __shared__ float logits[10];
    __shared__ float red[2];
    int g = blockIdx.x;
    int c = threadIdx.x;   // block = 128
    float cnt = fmaxf(gcnt[g], 1.0f);
    pooled[c] = gsum[g * 128 + c] / cnt;
    __syncthreads();
    if (c < 32) {
        float acc = fc1b[c];
        for (int k = 0; k < 128; ++k) acc += pooled[k] * fc1W[k * 32 + c];
        z[c] = fmaxf(acc, 0.0f);
    }
    __syncthreads();
    if (c < 10) {
        float acc = fc2b[c];
        for (int j = 0; j < 32; ++j) acc += z[j] * fc2W[j * 10 + c];
        logits[c] = acc;
    }
    __syncthreads();
    if (c == 0) {
        float mx = logits[0];
        for (int i = 1; i < 10; ++i) mx = fmaxf(mx, logits[i]);
        float s = 0.f;
        for (int i = 0; i < 10; ++i) s += __expf(logits[i] - mx);
        red[0] = mx; red[1] = __logf(s);
    }
    __syncthreads();
    if (c < 10) out[g * 10 + c] = logits[c] - red[0] - red[1];
}

// ---------------------------------------------------------------------------
// Host launcher
// ---------------------------------------------------------------------------
static inline unsigned nblk(long long n, int b) { return (unsigned)((n + b - 1) / b); }

extern "C" void kernel_launch(void* const* d_in, const int* in_sizes, int n_in,
                              void* d_out, int out_size, void* d_ws, size_t ws_size,
                              hipStream_t stream) {
    const float* x     = (const float*)d_in[0];
    const int*   ei    = (const int*)  d_in[1];
    const int*   batch = (const int*)  d_in[2];
    const float* W1    = (const float*)d_in[3];
    const float* a1s   = (const float*)d_in[4];
    const float* a1d   = (const float*)d_in[5];
    const float* b1    = (const float*)d_in[6];
    const float* W2    = (const float*)d_in[7];
    const float* a2s   = (const float*)d_in[8];
    const float* a2d   = (const float*)d_in[9];
    const float* b2    = (const float*)d_in[10];
    const float* W3    = (const float*)d_in[11];
    const float* a3s   = (const float*)d_in[12];
    const float* a3d   = (const float*)d_in[13];
    const float* b3    = (const float*)d_in[14];
    const float* fc1W  = (const float*)d_in[15];
    const float* fc1b  = (const float*)d_in[16];
    const float* fc2W  = (const float*)d_in[17];
    const float* fc2b  = (const float*)d_in[18];

    const int N    = in_sizes[2];        // 50000
    const int NE   = in_sizes[1] / 2;    // 800000
    const int Etot = NE + N;             // edges + self-loops

    // ---- workspace bump allocator (256B aligned) ----
    char* base = (char*)d_ws;
    size_t off = 0;
    auto alloc = [&](size_t bytes) -> void* {
        void* p = base + off;
        off += (bytes + 255) & ~(size_t)255;
        return p;
    };
    float*    hlin   = (float*)   alloc((size_t)N * 128 * 4);
    float*    haggr  = (float*)   alloc((size_t)N * 128 * 4);
    _Float16* hhalf  = (_Float16*)alloc((size_t)N * 128 * 2);
    float*    es     = (float*)   alloc((size_t)N * 8 * 4);
    float*    ed     = (float*)   alloc((size_t)N * 8 * 4);
    float*    mbuf   = (float*)   alloc((size_t)N * 8 * 4);
    float*    ssum   = (float*)   alloc((size_t)N * 8 * 4);
    float*    walpha = (float*)   alloc((size_t)Etot * 8 * 4);
    _Float16* wfrag  = (_Float16*)alloc((size_t)8 * 4 * 32 * 16 * 2);
    float*    gsum   = (float*)   alloc((size_t)NGRAPHS * 128 * 4);
    float*    gcnt   = (float*)   alloc((size_t)NGRAPHS * 4);
    (void)ws_size; (void)n_in; (void)out_size;

    const float NEGINF = -__builtin_inff();
    const int TB = 256;

    auto run_edges = [&](int C, int hcshift, const float* as_, const float* ad_) {
        const int HC = 1 << hcshift;
        const int cshift = (C == 16) ? 4 : 3;
        attn_coef_k<<<nblk((long long)N * 8, TB), TB, 0, stream>>>(hlin, as_, ad_, es, ed, N, C);
        fill_f32_k<<<nblk((long long)N * 8, TB), TB, 0, stream>>>(mbuf, NEGINF, N * 8);
        fill_f32_k<<<nblk((long long)N * 8, TB), TB, 0, stream>>>(ssum, 0.f, N * 8);
        fill_f32_k<<<nblk((long long)N * HC, TB), TB, 0, stream>>>(haggr, 0.f, N * HC);
        edge_max_k   <<<nblk((long long)Etot * 8, TB), TB, 0, stream>>>(ei, es, ed, mbuf, NE, Etot * 8);
        edge_expsum_k<<<nblk((long long)Etot * 8, TB), TB, 0, stream>>>(ei, es, ed, mbuf, ssum, walpha, NE, Etot * 8);
        edge_alpha_k <<<nblk((long long)Etot * 8, TB), TB, 0, stream>>>(ei, ssum, walpha, NE, Etot * 8);
        edge_aggr_k  <<<nblk((long long)Etot * HC, TB), TB, 0, stream>>>(ei, hlin, walpha, haggr,
                                                                         NE, hcshift, cshift, Etot * HC);
    };

    // ---------------- Layer 1 (K=2 -> scalar linear; HC=64, C=8) ----------------
    lin1_k<<<nblk((long long)N * 64, TB), TB, 0, stream>>>(x, W1, hlin, N);
    run_edges(8, 6, a1s, a1d);
    gat_finalize_k<<<nblk((long long)N * 64, TB), TB, 0, stream>>>(
        haggr, b1, nullptr, hhalf, 64, N * 64, 1);          // ELU, f16 for next WMMA

    // ---------------- Layer 2 (WMMA: K=64, M=128; HC=128, C=16) ----------------
    pack_wfrag_k<2><<<nblk(8 * 2 * 32 * 16, TB), TB, 0, stream>>>(W2, wfrag);
    gat_gemm_wmma_k<2><<<nblk(N, 64), 128, 0, stream>>>(hhalf, wfrag, hlin, N);
    run_edges(16, 7, a2s, a2d);
    gat_finalize_k<<<nblk((long long)N * 128, TB), TB, 0, stream>>>(
        haggr, b2, nullptr, hhalf, 128, N * 128, 1);        // ELU, f16 for next WMMA

    // ---------------- Layer 3 (WMMA: K=128, M=128; HC=128, C=16) ----------------
    pack_wfrag_k<4><<<nblk(8 * 4 * 32 * 16, TB), TB, 0, stream>>>(W3, wfrag);
    gat_gemm_wmma_k<4><<<nblk(N, 64), 128, 0, stream>>>(hhalf, wfrag, hlin, N);
    run_edges(16, 7, a3s, a3d);
    gat_finalize_k<<<nblk((long long)N * 128, TB), TB, 0, stream>>>(
        haggr, b3, haggr, nullptr, 128, N * 128, 0);        // no ELU, in-place f32

    // ---------------- Pool + head ----------------
    fill_f32_k<<<nblk(NGRAPHS * 128, TB), TB, 0, stream>>>(gsum, 0.f, NGRAPHS * 128);
    fill_f32_k<<<nblk(NGRAPHS, TB), TB, 0, stream>>>(gcnt, 0.f, NGRAPHS);
    pool_sum_k<<<nblk((long long)N * 128, TB), TB, 0, stream>>>(haggr, batch, gsum, N * 128);
    pool_cnt_k<<<nblk(N, TB), TB, 0, stream>>>(batch, gcnt, N);
    head_k<<<NGRAPHS, 128, 0, stream>>>(gsum, gcnt, fc1W, fc1b, fc2W, fc2b, (float*)d_out);
}
